// deltaor_47201690583534
// MI455X (gfx1250) — matrix operand order repaired and checked by reading, not compiled, not run on hardware
//
#include <hip/hip_runtime.h>
#include <hip/hip_bf16.h>

// ---------------------------------------------------------------------------
// Problem constants (B,N,M,D) = (8,2048,2048,2048).
// ---------------------------------------------------------------------------
#define BB 8
#define NN 2048
#define MM 2048
#define DD 2048
#define LN_EPS 1e-5f
#define SCALE 0.022097086912079608f   // 2048^-0.5

typedef __attribute__((ext_vector_type(16))) __bf16 v16bf;
typedef __attribute__((ext_vector_type(8)))  __bf16 v8bf;
typedef __attribute__((ext_vector_type(4)))  __bf16 v4bf;
typedef __attribute__((ext_vector_type(8)))  float  v8f;

// ---------------------------------------------------------------------------
// CDNA5 async global->LDS copy (16 bytes per lane), tracked by ASYNCcnt.
// LDS address = low 32 bits of the generic pointer (ISA 10.2: flat->LDS
// truncation; shared aperture base has zero low bits).
// ---------------------------------------------------------------------------
__device__ __forceinline__ void async_copy_b128(void* lds_dst, const void* gsrc) {
  unsigned ldsoff = (unsigned)(unsigned long long)lds_dst;
  asm volatile("global_load_async_to_lds_b128 %0, %1, off"
               :: "v"(ldsoff), "v"(gsrc) : "memory");
}
__device__ __forceinline__ void wait_async0() {
  asm volatile("s_wait_asynccnt 0" ::: "memory");
}

// ---------------------------------------------------------------------------
// 1) column mean of x over N  (two stage)
// ---------------------------------------------------------------------------
#define CCHUNKS 16
__global__ __launch_bounds__(256) void col_partial_kernel(
    const float* __restrict__ x, float* __restrict__ part) {
  const int b = blockIdx.z, ch = blockIdx.y;
  const int d = blockIdx.x * 256 + threadIdx.x;
  const int n0 = ch * (NN / CCHUNKS);
  const float* xb = x + (size_t)b * NN * DD;
  float s = 0.f;
  for (int n = n0; n < n0 + NN / CCHUNKS; ++n) s += xb[(size_t)n * DD + d];
  part[((size_t)b * CCHUNKS + ch) * DD + d] = s;
}

__global__ __launch_bounds__(256) void col_finish_kernel(
    const float* __restrict__ part, float* __restrict__ xmean) {
  const int b = blockIdx.y;
  const int d = blockIdx.x * 256 + threadIdx.x;
  float s = 0.f;
  for (int c = 0; c < CCHUNKS; ++c) s += part[((size_t)b * CCHUNKS + c) * DD + d];
  xmean[(size_t)b * DD + d] = s * (1.0f / NN);
}

// ---------------------------------------------------------------------------
// 2) per-row LayerNorm stats (mu, rstd); optional column-mean subtraction
// ---------------------------------------------------------------------------
__global__ __launch_bounds__(256) void row_stats_kernel(
    const float* __restrict__ X, const float* __restrict__ xmean,
    float2* __restrict__ st, int rowsPerBatch) {
  const size_t row = blockIdx.x;
  const float* xr = X + row * DD;
  const float* mr = xmean ? xmean + (row / rowsPerBatch) * (size_t)DD : nullptr;
  float s = 0.f, sq = 0.f;
  for (int c = threadIdx.x; c < DD; c += 256) {
    float v = xr[c] - (mr ? mr[c] : 0.f);
    s += v; sq += v * v;
  }
  __shared__ float rs[256], rq[256];
  rs[threadIdx.x] = s; rq[threadIdx.x] = sq;
  __syncthreads();
  for (int off = 128; off > 0; off >>= 1) {
    if (threadIdx.x < off) {
      rs[threadIdx.x] += rs[threadIdx.x + off];
      rq[threadIdx.x] += rq[threadIdx.x + off];
    }
    __syncthreads();
  }
  if (threadIdx.x == 0) {
    float mu = rs[0] * (1.0f / DD);
    float var = rq[0] * (1.0f / DD) - mu * mu;
    st[row] = make_float2(mu, rsqrtf(var + LN_EPS));
  }
}

// ---------------------------------------------------------------------------
// 3) normalize -> bf16   out = ((x - colmean?) - mu) * rstd
// ---------------------------------------------------------------------------
__global__ __launch_bounds__(256) void normalize_bf16_kernel(
    const float* __restrict__ X, const float* __restrict__ xmean,
    const float2* __restrict__ st, __bf16* __restrict__ O, int rowsPerBatch) {
  const size_t i = ((size_t)blockIdx.x * 256 + threadIdx.x) * 4;
  const size_t row = i / DD;
  const int col = (int)(i % DD);
  const float2 ms = st[row];
  float4 xv = *(const float4*)(X + i);
  float m0 = 0.f, m1 = 0.f, m2 = 0.f, m3 = 0.f;
  if (xmean) {
    const float* mr = xmean + (row / rowsPerBatch) * (size_t)DD + col;
    m0 = mr[0]; m1 = mr[1]; m2 = mr[2]; m3 = mr[3];
  }
  v4bf o;
  o[0] = (__bf16)((xv.x - m0 - ms.x) * ms.y);
  o[1] = (__bf16)((xv.y - m1 - ms.x) * ms.y);
  o[2] = (__bf16)((xv.z - m2 - ms.x) * ms.y);
  o[3] = (__bf16)((xv.w - m3 - ms.x) * ms.y);
  *(v4bf*)(O + i) = o;
}

__global__ __launch_bounds__(256) void f32_to_bf16_kernel(
    const float* __restrict__ X, __bf16* __restrict__ O) {
  const size_t i = ((size_t)blockIdx.x * 256 + threadIdx.x) * 4;
  float4 xv = *(const float4*)(X + i);
  v4bf o;
  o[0] = (__bf16)xv.x; o[1] = (__bf16)xv.y; o[2] = (__bf16)xv.z; o[3] = (__bf16)xv.w;
  *(v4bf*)(O + i) = o;
}

// ---------------------------------------------------------------------------
// 4) bf16 transpose (per batch):  in[rows][cols] -> out[cols][rows]
// ---------------------------------------------------------------------------
__global__ __launch_bounds__(256) void transpose_bf16_kernel(
    const __bf16* __restrict__ in, __bf16* __restrict__ out, int rows, int cols) {
  __shared__ __bf16 t[64][65];
  const int b = blockIdx.z;
  const __bf16* ib = in + (size_t)b * rows * cols;
  __bf16* ob = out + (size_t)b * rows * cols;
  const int r0 = blockIdx.y * 64, c0 = blockIdx.x * 64;
  for (int e = threadIdx.x; e < 64 * 64; e += 256) {
    int lr = e >> 6, lc = e & 63;
    t[lr][lc] = ib[(size_t)(r0 + lr) * cols + (c0 + lc)];
  }
  __syncthreads();
  for (int e = threadIdx.x; e < 64 * 64; e += 256) {
    int lc = e >> 6, lr = e & 63;
    ob[(size_t)(c0 + lc) * rows + (r0 + lr)] = t[lr][lc];
  }
}

// ---------------------------------------------------------------------------
// 5) WMMA NT GEMM:  C[m,n] = sum_k A[m,k]*B[n,k]   (bf16 in, f32 accum)
//    Block tile 128(M) x 256(N), BK=32, 256 threads = 8 waves (2x4),
//    wave tile 64x64 -> 16 v_wmma per wave per K-step.
//    Tile fill uses CDNA5 GLOBAL_LOAD_ASYNC_TO_LDS_B128 (ASYNCcnt).
//    MODE 0: C = bf16(acc)   MODE 1: C = f32(acc*scale)   MODE 2: C = f32(X+acc)
// ---------------------------------------------------------------------------
#define LDT 40  // LDS row stride in bf16 (32 + 8 pad; 80B, multiple of 16B)

__device__ __forceinline__ v16bf frag_a_ld(const __bf16* row0, int lane) {
  const int r = lane & 15, h = lane >> 4;
  const __bf16* p = row0 + (size_t)r * LDT;
  v8bf lo = *(const v8bf*)(p + 8 * h);        // K = 8h .. 8h+7
  v8bf hi = *(const v8bf*)(p + 16 + 8 * h);   // K = 16+8h .. 16+8h+7
  return __builtin_shufflevector(lo, hi, 0, 1, 2, 3, 4, 5, 6, 7,
                                 8, 9, 10, 11, 12, 13, 14, 15);
}

__device__ __forceinline__ v16bf frag_b_ld(const __bf16* row0, int lane) {
  const int n = lane & 15, h = lane >> 4;
  const __bf16* p = row0 + (size_t)n * LDT;
  v8bf lo = *(const v8bf*)(p + 8 * h);        // K = 8h .. 8h+7
  v8bf hi = *(const v8bf*)(p + 16 + 8 * h);   // K = 16+8h .. 16+8h+7
  return __builtin_shufflevector(lo, hi, 0, 1, 2, 3, 4, 5, 6, 7,
                                 8, 9, 10, 11, 12, 13, 14, 15);
}

template <int MODE>
__global__ __launch_bounds__(256) void gemm_nt_bf16_kernel(
    const __bf16* __restrict__ A, const __bf16* __restrict__ B,
    void* __restrict__ C, const float* __restrict__ X,
    int Kdim, int ldc,
    long long strideA, long long strideB, long long strideC, float scale) {
  const int bz = blockIdx.z;
  A += (size_t)bz * strideA;
  B += (size_t)bz * strideB;
  const size_t cOff = (size_t)bz * strideC;

  const int tid = threadIdx.x;
  const int lane = tid & 31;
  const int wave = tid >> 5;          // 0..7
  const int waveM = wave >> 2;        // 0..1  -> 64 rows each
  const int waveN = wave & 3;         // 0..3  -> 64 cols each
  const int blockM = blockIdx.y * 128;
  const int blockN = blockIdx.x * 256;

  __shared__ __bf16 As[128][LDT];   // 128 x 32 tile
  __shared__ __bf16 Bs[256][LDT];   // 256 x 32 tile

  v8f acc[4][4] = {};

  // async tile-fill indexing:
  //   A: 128 rows x 64B  -> 2 threads/row, 2 x 16B chunks each
  //   B: 256 rows x 64B  -> 1 thread/row,  4 x 16B chunks each
  const int arow = tid >> 1;
  const int acol = (tid & 1) * 16;    // bf16 elements

  for (int k0 = 0; k0 < Kdim; k0 += 32) {
    __syncthreads();  // previous iteration's reads are done
#pragma unroll
    for (int q = 0; q < 2; ++q)
      async_copy_b128(&As[arow][acol + q * 8],
                      A + (size_t)(blockM + arow) * Kdim + k0 + acol + q * 8);
#pragma unroll
    for (int q = 0; q < 4; ++q)
      async_copy_b128(&Bs[tid][q * 8],
                      B + (size_t)(blockN + tid) * Kdim + k0 + q * 8);
    wait_async0();
    __syncthreads();

    v16bf af[4];
#pragma unroll
    for (int mt = 0; mt < 4; ++mt)
      af[mt] = frag_a_ld(&As[waveM * 64 + mt * 16][0], lane);
#pragma unroll
    for (int nt = 0; nt < 4; ++nt) {
      v16bf bfv = frag_b_ld(&Bs[waveN * 64 + nt * 16][0], lane);
#pragma unroll
      for (int mt = 0; mt < 4; ++mt)
        acc[mt][nt] = __builtin_amdgcn_wmma_f32_16x16x32_bf16(
            false, af[mt], false, bfv, (short)0, acc[mt][nt], false, false);
    }
  }

  // epilogue:  lane -> (m = v + 8*(lane>>4), n = lane&15) within each 16x16
  const int h = lane >> 4, ln = lane & 15;
#pragma unroll
  for (int mt = 0; mt < 4; ++mt) {
#pragma unroll
    for (int nt = 0; nt < 4; ++nt) {
#pragma unroll
      for (int v = 0; v < 8; ++v) {
        const int r = blockM + waveM * 64 + mt * 16 + h * 8 + v;
        const int c = blockN + waveN * 64 + nt * 16 + ln;
        const size_t idx = cOff + (size_t)r * ldc + c;
        const float val = acc[mt][nt][v];
        if (MODE == 0) ((__bf16*)C)[idx] = (__bf16)val;
        if (MODE == 1) ((float*)C)[idx] = val * scale;
        if (MODE == 2) ((float*)C)[idx] = X[idx] + val;
      }
    }
  }
}

// ---------------------------------------------------------------------------
// 6) row softmax over M, output bf16 probabilities
// ---------------------------------------------------------------------------
__global__ __launch_bounds__(256) void softmax_rows_kernel(
    const float* __restrict__ S, __bf16* __restrict__ P) {
  const size_t row = blockIdx.x;
  const float* s = S + row * MM;
  __bf16* p = P + row * MM;
  __shared__ float red[256];

  float lmax = -3.4e38f;
  for (int c = threadIdx.x; c < MM; c += 256) lmax = fmaxf(lmax, s[c]);
  red[threadIdx.x] = lmax;
  __syncthreads();
  for (int off = 128; off > 0; off >>= 1) {
    if (threadIdx.x < off)
      red[threadIdx.x] = fmaxf(red[threadIdx.x], red[threadIdx.x + off]);
    __syncthreads();
  }
  const float m = red[0];
  __syncthreads();

  float lsum = 0.f;
  for (int c = threadIdx.x; c < MM; c += 256) lsum += __expf(s[c] - m);
  red[threadIdx.x] = lsum;
  __syncthreads();
  for (int off = 128; off > 0; off >>= 1) {
    if (threadIdx.x < off) red[threadIdx.x] += red[threadIdx.x + off];
    __syncthreads();
  }
  const float inv = 1.0f / red[0];
  for (int c = threadIdx.x; c < MM; c += 256)
    p[c] = (__bf16)(__expf(s[c] - m) * inv);
}

// ---------------------------------------------------------------------------
// launch
// ---------------------------------------------------------------------------
extern "C" void kernel_launch(void* const* d_in, const int* in_sizes, int n_in,
                              void* d_out, int out_size, void* d_ws, size_t ws_size,
                              hipStream_t stream) {
  const float* x  = (const float*)d_in[0];
  const float* g  = (const float*)d_in[1];
  const float* Wq = (const float*)d_in[2];
  const float* Wg = (const float*)d_in[3];
  float* out = (float*)d_out;
  char* ws = (char*)d_ws;

  // workspace layout (bytes)
  const size_t SZ_BND = (size_t)BB * NN * DD;  // == B*M*D == B*N*M
  __bf16* q_bf   = (__bf16*)(ws);                           // 64 MB
  __bf16* gc_bf  = (__bf16*)(ws + 1 * SZ_BND * 2);          // 64 MB
  __bf16* gcT_bf = (__bf16*)(ws + 2 * SZ_BND * 2);          // 64 MB
  __bf16* rq_bf  = (__bf16*)(ws + 3 * SZ_BND * 2);          // 64 MB
  __bf16* rg_bf  = (__bf16*)(ws + 4 * SZ_BND * 2);          // 64 MB
  __bf16* P_bf   = (__bf16*)(ws + 5 * SZ_BND * 2);          // 64 MB
  float*  S_f    = (float*) (ws + 6 * SZ_BND * 2);          // 128 MB
  __bf16* Wq_bf  = (__bf16*)(ws + 6 * SZ_BND * 2 + SZ_BND * 4);
  __bf16* Wg_bf  = (__bf16*)(ws + 6 * SZ_BND * 2 + SZ_BND * 4 + (size_t)DD * DD * 2);
  float*  xmean  = (float*) (ws + 6 * SZ_BND * 2 + SZ_BND * 4 + (size_t)DD * DD * 4);
  float*  part   = xmean + (size_t)BB * DD;
  float2* st_x   = (float2*)(part + (size_t)BB * CCHUNKS * DD);
  float2* st_g   = st_x + (size_t)BB * NN;

  // 1) column mean of x over N
  col_partial_kernel<<<dim3(DD / 256, CCHUNKS, BB), 256, 0, stream>>>(x, part);
  col_finish_kernel<<<dim3(DD / 256, BB), 256, 0, stream>>>(part, xmean);

  // 2) row stats
  row_stats_kernel<<<dim3(BB * NN), 256, 0, stream>>>(x, nullptr, st_x, NN);
  row_stats_kernel<<<dim3(BB * MM), 256, 0, stream>>>(g, xmean, st_g, MM);

  // 3) normalize to bf16 + weight conversion
  normalize_bf16_kernel<<<dim3((unsigned)(SZ_BND / 4 / 256)), 256, 0, stream>>>(
      x, nullptr, st_x, q_bf, NN);
  normalize_bf16_kernel<<<dim3((unsigned)(SZ_BND / 4 / 256)), 256, 0, stream>>>(
      g, xmean, st_g, gc_bf, MM);
  f32_to_bf16_kernel<<<dim3((DD * DD / 4) / 256), 256, 0, stream>>>(Wq, Wq_bf);
  f32_to_bf16_kernel<<<dim3((DD * DD / 4) / 256), 256, 0, stream>>>(Wg, Wg_bf);

  // 4) gc transpose per batch: [M][D] -> [D][M]
  transpose_bf16_kernel<<<dim3(DD / 64, MM / 64, BB), 256, 0, stream>>>(
      gc_bf, gcT_bf, MM, DD);

  // 5) rel_q = q @ Wq^T   (single [B*N x D] x [D x D]^T GEMM)
  gemm_nt_bf16_kernel<0><<<dim3(DD / 256, (BB * NN) / 128, 1), 256, 0, stream>>>(
      q_bf, Wq_bf, rq_bf, nullptr, DD, DD, 0, 0, 0, 1.0f);
  //    rel_g = gc @ Wg^T
  gemm_nt_bf16_kernel<0><<<dim3(DD / 256, (BB * MM) / 128, 1), 256, 0, stream>>>(
      gc_bf, Wg_bf, rg_bf, nullptr, DD, DD, 0, 0, 0, 1.0f);

  // 6) scores S = (rel_q @ rel_g^T) * SCALE   (batched)
  gemm_nt_bf16_kernel<1><<<dim3(MM / 256, NN / 128, BB), 256, 0, stream>>>(
      rq_bf, rg_bf, S_f, nullptr, DD, MM,
      (long long)NN * DD, (long long)MM * DD, (long long)NN * MM, SCALE);

  // 7) softmax over M -> bf16 P
  softmax_rows_kernel<<<dim3(BB * NN), 256, 0, stream>>>(S_f, P_bf);

  // 8) out = x + P @ gcT^T   (batched, K = M)
  gemm_nt_bf16_kernel<2><<<dim3(DD / 256, NN / 128, BB), 256, 0, stream>>>(
      P_bf, gcT_bf, out, x, MM, DD,
      (long long)NN * MM, (long long)DD * MM, (long long)NN * DD, 1.0f);
}
